// RNNTwoLayerModel_47682726921063
// MI455X (gfx1250) — matrix operand order repaired and checked by reading, not compiled, not run on hardware
//
#include <hip/hip_runtime.h>
#include <hip/hip_bf16.h>
#include <math.h>

#define SEQ   1024
#define EMB   256
#define HID   512
#define OUTD  128
#define BATCH 64

// LDS row paddings (in bf16 elements) to break bank conflicts on A-fragment loads
#define EPAD  (EMB + 8)    // 264 -> 528B rows (16B aligned)
#define H1PAD (HID + 8)    // 520 -> 1040B rows
#define H2PAD (OUTD + 8)   // 136 -> 272B rows

#define E_BYTES  (BATCH * EPAD * 2)     // one E buffer  (33,792B)
#define H1_BYTES (BATCH * H1PAD * 2)    // one H1 buffer (66,560B)
#define H2_BYTES (BATCH * H2PAD * 2)    // one H2 buffer (17,408B)
// 2xE + 2xH1 + 2xH2 = 235,520B  (< 320KB WGP LDS)
#define SMEM_BYTES (2 * E_BYTES + 2 * H1_BYTES + 2 * H2_BYTES)

typedef __attribute__((ext_vector_type(16))) __bf16 v16bf;
typedef __attribute__((ext_vector_type(8)))  float  v8f;

#if __has_builtin(__builtin_amdgcn_tanhf)
#define fast_tanh(x) __builtin_amdgcn_tanhf(x)
#else
#define fast_tanh(x) tanhf(x)
#endif

__device__ __forceinline__ unsigned short f2bf(float f) {
    unsigned int u = __float_as_uint(f);
    unsigned int r = u + 0x7FFFu + ((u >> 16) & 1u);   // round-to-nearest-even
    return (unsigned short)(r >> 16);
}
__device__ __forceinline__ unsigned int pack2(float a, float b) {
    return (unsigned int)f2bf(a) | ((unsigned int)f2bf(b) << 16);
}

// Fresh SGPR zero each call: adding it to a pointer blocks loop-invariant
// hoisting of derived loads (round 1 spilled 44 hoisted fragments to scratch)
// while PRESERVING address-space provenance so loads stay global_load (round 2
// pointer-laundering degraded them to flat_load, coupling LOADcnt with DScnt).
__device__ __forceinline__ int opaque_zero() {
    int z = 0;
    asm volatile("" : "+s"(z));
    return z;
}

// Async gather: copy 32B from global (bf16 emb row) straight into LDS.
// VDST VGPR = LDS byte offset; immediate offset advances BOTH addresses.
__device__ __forceinline__ void async_copy32(unsigned lds_off, unsigned long long gaddr) {
    asm volatile("global_load_async_to_lds_b128 %0, %1, off"
                 :: "v"(lds_off), "v"(gaddr) : "memory");
    asm volatile("global_load_async_to_lds_b128 %0, %1, off offset:16"
                 :: "v"(lds_off), "v"(gaddr) : "memory");
}
__device__ __forceinline__ void wait_async0() {
#if __has_builtin(__builtin_amdgcn_s_wait_asynccnt)
    __builtin_amdgcn_s_wait_asynccnt(0);
#else
    asm volatile("s_wait_asynccnt 0x0" ::: "memory");
#endif
}

// ---- fragment loaders -------------------------------------------------------
// A (16x32 bf16, row-major in LDS). Lane l: m = l&15, half = l>>4.
__device__ __forceinline__ v16bf ldA(const unsigned short* row, int ktByte, int half) {
    union { uint4 q[2]; v16bf v; } u;
    const char* p = (const char*)row + ktByte + half * 16;
    u.q[0] = *(const uint4*)(p);
    u.q[1] = *(const uint4*)(p + 32);
    return u.v;
}
// B fragments pre-packed in workspace: tile -> 32 lanes x 8 dwords contiguous.
__device__ __forceinline__ v16bf ldB(const unsigned int* __restrict__ P, int tile, int lane) {
    union { uint4 q[2]; v16bf v; } u;
    const uint4* p = (const uint4*)(P + ((size_t)tile * 32 + lane) * 8);
    u.q[0] = p[0];
    u.q[1] = p[1];
    return u.v;
}
__device__ __forceinline__ v8f wmma_bf16(v16bf a, v16bf b, v8f c) {
    return __builtin_amdgcn_wmma_f32_16x16x32_bf16(false, a, false, b, (short)0, c, false, false);
}

// ---- prep: pack W [N,K] f32 -> bf16 WMMA-B fragments ------------------------
__global__ void pack_w_kernel(const float* __restrict__ W, int N, int K,
                              unsigned int* __restrict__ P) {
    int NT = N >> 4, KT = K >> 5;
    int g = blockIdx.x * blockDim.x + threadIdx.x;   // ((kt*NT + nt)*32 + lane)
    if (g >= KT * NT * 32) return;
    int lane = g & 31;
    int tile = g >> 5;
    int nt = tile % NT, kt = tile / NT;
    int n = nt * 16 + (lane & 15);
    int khalf = lane >> 4;
    const float* row = W + (size_t)n * K + kt * 32 + khalf * 8;
    unsigned int d[8];
#pragma unroll
    for (int v = 0; v < 8; ++v) d[v] = pack2(row[2 * v + (v >= 4 ? 8 : 0)], row[2 * v + 1 + (v >= 4 ? 8 : 0)]);
    uint4* dst = (uint4*)(P + (size_t)g * 8);
    dst[0] = make_uint4(d[0], d[1], d[2], d[3]);
    dst[1] = make_uint4(d[4], d[5], d[6], d[7]);
}

// ---- prep: embedding table f32 -> bf16 (pairs packed into dwords) -----------
__global__ void emb_bf_kernel(const float* __restrict__ E, unsigned int* __restrict__ O, int n) {
    int g = blockIdx.x * blockDim.x + threadIdx.x;
    if (g >= n) return;
    float2 f = ((const float2*)E)[g];
    O[g] = pack2(f.x, f.y);
}

__global__ void bias_kernel(const float* __restrict__ bi1, const float* __restrict__ bh1,
                            const float* __restrict__ bi2, const float* __restrict__ bh2,
                            float* __restrict__ o1, float* __restrict__ o2) {
    int i = blockIdx.x * blockDim.x + threadIdx.x;
    if (i < HID)  o1[i] = bi1[i] + bh1[i];
    if (i < OUTD) o2[i] = bi2[i] + bh2[i];
}

// ---- persistent RNN: one workgroup, 32 waves, whole scan --------------------
__global__ void __launch_bounds__(1024)
rnn_kernel(const int* __restrict__ xs, const unsigned short* __restrict__ embbf,
           const unsigned int* __restrict__ Pih1g, const unsigned int* __restrict__ Phh1g,
           const unsigned int* __restrict__ Pih2g, const unsigned int* __restrict__ Phh2g,
           const float* __restrict__ bias1g, const float* __restrict__ bias2g,
           float* __restrict__ out) {
    extern __shared__ unsigned char smem_raw[];
    unsigned short* Ebf = (unsigned short*)smem_raw;                    // [2][64][EPAD]  (LDS off 0)
    unsigned short* H1  = (unsigned short*)(smem_raw + 2 * E_BYTES);    // [2][64][H1PAD]
    unsigned short* H2  = (unsigned short*)(smem_raw + 2 * E_BYTES + 2 * H1_BYTES); // [2][64][H2PAD]

    const int tid  = threadIdx.x;
    const int wv   = tid >> 5;
    const int lane = tid & 31;
    const int lh   = lane >> 4;      // half
    const int ln   = lane & 15;

    // zero initial hidden state (both parities)
    for (int i = tid; i < 2 * BATCH * H1PAD; i += 1024) H1[i] = 0;
    for (int i = tid; i < 2 * BATCH * H2PAD; i += 1024) H2[i] = 0;

    const int gb = tid >> 4;         // 0..63 batch row for gather
    const int gc = tid & 15;         // 32B chunk within the 512B bf16 row
    const unsigned long long embBase = (unsigned long long)(size_t)embbf;

    // prologue: async-gather E for t=0 into buffer 0
    {
        int idx = xs[gb * SEQ + 0];
        async_copy32((unsigned)(gb * (EPAD * 2) + gc * 32),
                     embBase + (unsigned long long)idx * (EMB * 2) + gc * 32);
    }
    wait_async0();
    __syncthreads();

    for (int t = 0; t < SEQ; ++t) {
        const int zo = opaque_zero();            // defeat cross-step hoisting
        const unsigned int* Pih1 = Pih1g + zo;
        const unsigned int* Phh1 = Phh1g + zo;
        const unsigned int* Pih2 = Pih2g + zo;
        const unsigned int* Phh2 = Phh2g + zo;
        const float* bias1 = bias1g + zo;
        const float* bias2 = bias2g + zo;

        const int pr = t & 1, pw = pr ^ 1;
        const unsigned short* Ecur = Ebf + pr * BATCH * EPAD;
        unsigned short* H1r = H1 + pr * BATCH * H1PAD;
        unsigned short* H1w = H1 + pw * BATCH * H1PAD;
        unsigned short* H2r = H2 + pr * BATCH * H2PAD;
        unsigned short* H2w = H2 + pw * BATCH * H2PAD;

        // ---- kick off async gather of E_{t+1} into the other buffer;
        //      completes under the compute below, waited at loop bottom.
        if (t + 1 < SEQ) {
            int idx = xs[gb * SEQ + t + 1];
            async_copy32((unsigned)(pw * E_BYTES + gb * (EPAD * 2) + gc * 32),
                         embBase + (unsigned long long)idx * (EMB * 2) + gc * 32);
        }

        // ---- layer 1: h1 = tanh(E_t @ Wih1^T + h1_prev @ Whh1^T + b1)
        // wave wv owns N-tile nt = wv (16 cols), all 4 M-tiles.
        {
            const int nt = wv;
            v8f acc0 = {0,0,0,0,0,0,0,0}, acc1 = acc0, acc2 = acc0, acc3 = acc0;

            for (int kt = 0; kt < EMB / 32; ++kt) {
                v16bf bf = ldB(Pih1, kt * 32 + nt, lane);
                if (kt + 1 < EMB / 32)
                    __builtin_prefetch(Pih1 + ((size_t)((kt + 1) * 32 + nt) * 32 + lane) * 8, 0, 3);
                int kb = kt * 64;
                acc0 = wmma_bf16(ldA(Ecur + (0 * 16 + ln) * EPAD, kb, lh), bf, acc0);
                acc1 = wmma_bf16(ldA(Ecur + (1 * 16 + ln) * EPAD, kb, lh), bf, acc1);
                acc2 = wmma_bf16(ldA(Ecur + (2 * 16 + ln) * EPAD, kb, lh), bf, acc2);
                acc3 = wmma_bf16(ldA(Ecur + (3 * 16 + ln) * EPAD, kb, lh), bf, acc3);
            }
            for (int kt = 0; kt < HID / 32; ++kt) {
                v16bf bf = ldB(Phh1, kt * 32 + nt, lane);
                if (kt + 1 < HID / 32)
                    __builtin_prefetch(Phh1 + ((size_t)((kt + 1) * 32 + nt) * 32 + lane) * 8, 0, 3);
                int kb = kt * 64;
                acc0 = wmma_bf16(ldA(H1r + (0 * 16 + ln) * H1PAD, kb, lh), bf, acc0);
                acc1 = wmma_bf16(ldA(H1r + (1 * 16 + ln) * H1PAD, kb, lh), bf, acc1);
                acc2 = wmma_bf16(ldA(H1r + (2 * 16 + ln) * H1PAD, kb, lh), bf, acc2);
                acc3 = wmma_bf16(ldA(H1r + (3 * 16 + ln) * H1PAD, kb, lh), bf, acc3);
            }
            float bn = bias1[nt * 16 + ln];
            v8f acc[4] = {acc0, acc1, acc2, acc3};
#pragma unroll
            for (int mt = 0; mt < 4; ++mt)
#pragma unroll
                for (int v = 0; v < 8; ++v) {
                    float xv = fast_tanh(acc[mt][v] + bn);
                    H1w[(mt * 16 + lh * 8 + v) * H1PAD + nt * 16 + ln] = f2bf(xv);
                }
        }
        __syncthreads();

        // ---- layer 2: h2 = tanh(h1_t @ Wih2^T + h2_prev @ Whh2^T + b2)
        // 32 tiles (4 M x 8 N) -> one per wave.
        {
            const int mt2 = wv >> 3, nt2 = wv & 7;
            v8f acc = {0,0,0,0,0,0,0,0};
            const unsigned short* arow1 = H1w + (mt2 * 16 + ln) * H1PAD;
            for (int kt = 0; kt < HID / 32; ++kt) {
                v16bf bf = ldB(Pih2, kt * 8 + nt2, lane);
                if (kt + 1 < HID / 32)
                    __builtin_prefetch(Pih2 + ((size_t)((kt + 1) * 8 + nt2) * 32 + lane) * 8, 0, 3);
                acc = wmma_bf16(ldA(arow1, kt * 64, lh), bf, acc);
            }
            const unsigned short* arow2 = H2r + (mt2 * 16 + ln) * H2PAD;
            for (int kt = 0; kt < OUTD / 32; ++kt) {
                v16bf bf = ldB(Phh2, kt * 8 + nt2, lane);
                acc = wmma_bf16(ldA(arow2, kt * 64, lh), bf, acc);
            }
            float bn = bias2[nt2 * 16 + ln];
#pragma unroll
            for (int v = 0; v < 8; ++v) {
                float xv = fast_tanh(acc[v] + bn);
                int bb = mt2 * 16 + lh * 8 + v;
                int oo = nt2 * 16 + ln;
                H2w[bb * H2PAD + oo] = f2bf(xv);
                out[((size_t)bb * OUTD + oo) * SEQ + t] = xv;   // [B][OUT][T]
            }
        }
        wait_async0();        // E_{t+1} landed (hidden under layer1+layer2)
        __syncthreads();
    }
}

// ---- softmax over OUT axis of [B][OUT][T], in place; one wave per (b,t) ----
__global__ void softmax_kernel(float* __restrict__ out) {
    int tid  = threadIdx.x;
    int lane = tid & 31;
    int w    = blockIdx.x * (blockDim.x >> 5) + (tid >> 5);
    int b    = w >> 10;        // / SEQ
    int t    = w & (SEQ - 1);
    float* base = out + ((size_t)b * OUTD) * SEQ + t;
    float v[4];
#pragma unroll
    for (int i = 0; i < 4; ++i) v[i] = base[(size_t)(lane + 32 * i) * SEQ];
    float m = fmaxf(fmaxf(v[0], v[1]), fmaxf(v[2], v[3]));
    for (int off = 16; off; off >>= 1) m = fmaxf(m, __shfl_xor(m, off, 32));
    float s = 0.f;
#pragma unroll
    for (int i = 0; i < 4; ++i) { v[i] = __expf(v[i] - m); s += v[i]; }
    for (int off = 16; off; off >>= 1) s += __shfl_xor(s, off, 32);
    float inv = 1.0f / s;
#pragma unroll
    for (int i = 0; i < 4; ++i) base[(size_t)(lane + 32 * i) * SEQ] = v[i] * inv;
}

extern "C" void kernel_launch(void* const* d_in, const int* in_sizes, int n_in,
                              void* d_out, int out_size, void* d_ws, size_t ws_size,
                              hipStream_t stream) {
    (void)in_sizes; (void)n_in; (void)out_size; (void)ws_size;
    const int*   xs   = (const int*)d_in[0];
    const float* emb  = (const float*)d_in[1];
    const float* Wih1 = (const float*)d_in[2];
    const float* Whh1 = (const float*)d_in[3];
    const float* bih1 = (const float*)d_in[4];
    const float* bhh1 = (const float*)d_in[5];
    const float* Wih2 = (const float*)d_in[6];
    const float* Whh2 = (const float*)d_in[7];
    const float* bih2 = (const float*)d_in[8];
    const float* bhh2 = (const float*)d_in[9];
    float* out = (float*)d_out;

    // workspace layout (dwords)
    unsigned int* ws   = (unsigned int*)d_ws;
    unsigned int* Pih1 = ws;                                    // 65536 dwords
    unsigned int* Phh1 = Pih1 + (EMB/32)*(HID/16)*32*8;         // 131072
    unsigned int* Pih2 = Phh1 + (HID/32)*(HID/16)*32*8;         // 32768
    unsigned int* Phh2 = Pih2 + (HID/32)*(OUTD/16)*32*8;        // 8192
    float* bias1 = (float*)(Phh2 + (OUTD/32)*(OUTD/16)*32*8);
    float* bias2 = bias1 + HID;
    unsigned int* embbf = (unsigned int*)(bias2 + OUTD);        // 32000*256 bf16 = 16MB
    const int embDwords = 32000 * EMB / 2;                      // 4,096,000

    pack_w_kernel<<<(EMB/32)*(HID/16)*32/256,  256, 0, stream>>>(Wih1, HID,  EMB,  Pih1);
    pack_w_kernel<<<(HID/32)*(HID/16)*32/256,  256, 0, stream>>>(Whh1, HID,  HID,  Phh1);
    pack_w_kernel<<<(HID/32)*(OUTD/16)*32/256, 256, 0, stream>>>(Wih2, OUTD, HID,  Pih2);
    pack_w_kernel<<<(OUTD/32)*(OUTD/16)*32/256,256, 0, stream>>>(Whh2, OUTD, OUTD, Phh2);
    emb_bf_kernel<<<embDwords / 256, 256, 0, stream>>>(emb, embbf, embDwords);
    bias_kernel<<<2, 256, 0, stream>>>(bih1, bhh1, bih2, bhh2, bias1, bias2);

    rnn_kernel<<<1, 1024, SMEM_BYTES, stream>>>(xs, (const unsigned short*)embbf,
                                                Pih1, Phh1, Pih2, Phh2,
                                                bias1, bias2, out);

    softmax_kernel<<<BATCH * SEQ / 8, 256, 0, stream>>>(out);
}
